// Adapt_Layer_3513283248270
// MI455X (gfx1250) — matrix operand order repaired and checked by reading, not compiled, not run on hardware
//
#include <hip/hip_runtime.h>

#define Dd   1024
#define Cc   8
#define Bb   16384
#define KK   8192        // C*D virtual K
#define BM   128
#define BN   128
#define BK   32
#define TB   256

typedef __attribute__((ext_vector_type(16))) __bf16 v16bf;
typedef __attribute__((ext_vector_type(2)))  __bf16 v2bf;
typedef __attribute__((ext_vector_type(8)))  float  v8f;

union FragU { uint4 u4[2]; v16bf v; };

// f32x4 -> packed bf16 hi (RNE) + bf16 lo (residual), via hardware cvt ops
__device__ __forceinline__ void split4(const float4 f, uint2& hi, uint2& lo) {
    __bf16 h0 = (__bf16)f.x, h1 = (__bf16)f.y, h2 = (__bf16)f.z, h3 = (__bf16)f.w;
    v2bf p01 = {h0, h1}, p23 = {h2, h3};
    hi.x = __builtin_bit_cast(unsigned, p01);
    hi.y = __builtin_bit_cast(unsigned, p23);
    float r0 = f.x - (float)h0, r1 = f.y - (float)h1;
    float r2 = f.z - (float)h2, r3 = f.w - (float)h3;
    v2bf q01 = {(__bf16)r0, (__bf16)r1}, q23 = {(__bf16)r2, (__bf16)r3};
    lo.x = __builtin_bit_cast(unsigned, q01);
    lo.y = __builtin_bit_cast(unsigned, q23);
}

// ---------------------------------------------------------------------------
// Kernel 0: one-shot split of W (f32) into packed-bf16 hi/lo planes in ws.
// ---------------------------------------------------------------------------
__global__ __launch_bounds__(256) void wsplit(
    const float* __restrict__ W, uint2* __restrict__ Wh, uint2* __restrict__ Wl)
{
    const size_t i = (size_t)blockIdx.x * 256 + threadIdx.x;   // one float4 each
    const float4 f = ((const float4*)W)[i];
    uint2 h, l;
    split4(f, h, l);
    Wh[i] = h;
    Wl[i] = l;
}

// ---------------------------------------------------------------------------
// Kernel 1: fused prob-weighted expert GEMM
//   pred[b,o] = sum_{k} (prob[b,k>>10]*feature[b,k&1023]) * W[k>>10, o, k&1023]
// fp32 emulated via bf16 hi/lo split: Ah*Bh + Ah*Bl + Al*Bh
// ---------------------------------------------------------------------------
__global__ __launch_bounds__(TB) void expert_gemm(
    const float* __restrict__ feat, const float* __restrict__ prob,
    const uint2* __restrict__ Wh, const uint2* __restrict__ Wl,
    float* __restrict__ out)
{
    // LDS tiles: row pitch = 5 uint4 = 80B (bank-conflict-free fragment loads)
    __shared__ uint4 sAh[2][BM * 5];
    __shared__ uint4 sAl[2][BM * 5];
    __shared__ uint4 sBh[2][BM * 5];
    __shared__ uint4 sBl[2][BM * 5];

    const int t    = threadIdx.x;
    const int bm   = blockIdx.x;           // 0..127  M tiles
    const int bn   = blockIdx.y;           // 0..7    N tiles
    const int lane = t & 31;
    const int wave = t >> 5;
    const int wm   = (wave >> 2) * 64;     // wave M offset
    const int wn   = (wave & 3) * 32;      // wave N offset
    const int half = lane >> 4;
    const int l15  = lane & 15;

    // staging: 8 threads per row (float4 / uint2-pair each), 32 rows/pass, 4 passes
    const int col4 = t & 7;
    const int row0 = t >> 3;

    const int arow = bm * BM;
    const int brow = bn * BN;

    v8f acc[4][2] = {};

    float4 ga[4];
    float  gp[4];
    uint2  gbh[4], gbl[4];

    auto load_global = [&](int kb) {
        const int k0 = kb * BK;
        const int c  = k0 >> 10;           // expert index (constant within K-block)
        const int kc = k0 & (Dd - 1);
        const float* fp = feat + (size_t)arow * Dd + kc + col4 * 4;
        const size_t eb = (size_t)c * Dd * Dd + (size_t)brow * Dd + kc + col4 * 4;
#pragma unroll
        for (int p = 0; p < 4; ++p) {
            const int r = row0 + 32 * p;
            ga[p]  = *(const float4*)(fp + (size_t)r * Dd);
            gp[p]  = prob[(size_t)(arow + r) * Cc + c];
            const size_t e = (eb + (size_t)r * Dd) >> 2;   // uint2 = 4 bf16
            gbh[p] = Wh[e];
            gbl[p] = Wl[e];
        }
    };

    auto store_lds = [&](int buf) {
#pragma unroll
        for (int p = 0; p < 4; ++p) {
            const int r = row0 + 32 * p;
            // A side: prob * feature, hardware split hi/lo
            const float s = gp[p];
            float4 x = make_float4(ga[p].x * s, ga[p].y * s, ga[p].z * s, ga[p].w * s);
            uint2 hv, lv;
            split4(x, hv, lv);
            ((uint2*)sAh[buf])[r * 10 + col4] = hv;
            ((uint2*)sAl[buf])[r * 10 + col4] = lv;
            // B side: pre-split bf16, straight copy
            ((uint2*)sBh[buf])[r * 10 + col4] = gbh[p];
            ((uint2*)sBl[buf])[r * 10 + col4] = gbl[p];
        }
    };

    auto compute = [&](int buf) {
        // B fragments (K x N: lanes 0-15 -> K=0..7,16..23, lanes 16-31 -> K=8..15,24..31)
        FragU bh[2], bl[2];
#pragma unroll
        for (int ni = 0; ni < 2; ++ni) {
            const int nr = (wn + ni * 16 + l15) * 5 + half;
            bh[ni].u4[0] = sBh[buf][nr];  bh[ni].u4[1] = sBh[buf][nr + 2];
            bl[ni].u4[0] = sBl[buf][nr];  bl[ni].u4[1] = sBl[buf][nr + 2];
        }
        // software-pipelined A fragments: load mi+1 while computing mi
        FragU ah, al, ah2, al2;
        {
            const int mr = (wm + l15) * 5 + half;
            ah.u4[0] = sAh[buf][mr];  ah.u4[1] = sAh[buf][mr + 2];
            al.u4[0] = sAl[buf][mr];  al.u4[1] = sAl[buf][mr + 2];
        }
#pragma unroll
        for (int mi = 0; mi < 4; ++mi) {
            if (mi < 3) {
                const int mr = (wm + (mi + 1) * 16 + l15) * 5 + half;
                ah2.u4[0] = sAh[buf][mr];  ah2.u4[1] = sAh[buf][mr + 2];
                al2.u4[0] = sAl[buf][mr];  al2.u4[1] = sAl[buf][mr + 2];
            }
#pragma unroll
            for (int ni = 0; ni < 2; ++ni) {
                acc[mi][ni] = __builtin_amdgcn_wmma_f32_16x16x32_bf16(
                    false, ah.v, false, bh[ni].v, (short)0, acc[mi][ni], false, false);
                acc[mi][ni] = __builtin_amdgcn_wmma_f32_16x16x32_bf16(
                    false, ah.v, false, bl[ni].v, (short)0, acc[mi][ni], false, false);
                acc[mi][ni] = __builtin_amdgcn_wmma_f32_16x16x32_bf16(
                    false, al.v, false, bh[ni].v, (short)0, acc[mi][ni], false, false);
            }
            ah = ah2;  al = al2;   // rotation; dissolves under full unroll
        }
    };

    const int NKB = KK / BK;   // 256
    load_global(0);
    store_lds(0);
    __syncthreads();
    for (int kb = 0; kb < NKB; ++kb) {
        if (kb + 1 < NKB) load_global(kb + 1);
        compute(kb & 1);
        if (kb + 1 < NKB) store_lds((kb + 1) & 1);   // other buffer: no hazard
        __syncthreads();
    }

    // C/D layout: lanes 0-15: M=v, lanes 16-31: M=v+8, N = lane&15
#pragma unroll
    for (int mi = 0; mi < 4; ++mi)
#pragma unroll
        for (int ni = 0; ni < 2; ++ni) {
            const int grow0 = bm * BM + wm + mi * 16 + half * 8;
            const int gcol  = bn * BN + wn + ni * 16 + l15;
#pragma unroll
            for (int v = 0; v < 8; ++v)
                out[(size_t)(grow0 + v) * Dd + gcol] = acc[mi][ni][v];
        }
}

// ---------------------------------------------------------------------------
// Kernel 2: bias + gating epilogue. One wave per row; reads pred from `out`
// (written by kernel 1) and rewrites it in place (row exclusively owned).
// ---------------------------------------------------------------------------
__global__ __launch_bounds__(256) void gate_epilogue(
    float* __restrict__ out, const float* __restrict__ feat,
    const float* __restrict__ prob, const float* __restrict__ bias,
    const float* __restrict__ pw_w, const float* __restrict__ pw_b,
    const float* __restrict__ fw_w, const float* __restrict__ fw_b)
{
    __shared__ float sb[Cc * Dd];       // 32KB bias cache
    const int t = threadIdx.x;
    for (int i = t; i < Cc * Dd; i += 256) sb[i] = bias[i];
    __syncthreads();

    const int lane = t & 31;
    const int wave = t >> 5;
    const int b = blockIdx.x * 8 + wave;

    float pc[Cc];
#pragma unroll
    for (int c = 0; c < Cc; ++c) pc[c] = prob[(size_t)b * Cc + c];

    const float* pr = out + (size_t)b * Dd;
    const float* fr = feat + (size_t)b * Dd;

    float predv[32], featv[32];
    float sp = 0.f, sf = 0.f;
#pragma unroll
    for (int j = 0; j < 32; ++j) {
        const int o = j * 32 + lane;
        float p = pr[o];
#pragma unroll
        for (int c = 0; c < Cc; ++c) p += pc[c] * sb[c * Dd + o];
        const float f = fr[o];
        predv[j] = p; featv[j] = f;
        sp += p * pw_w[o];
        sf += f * fw_w[o];
    }
#pragma unroll
    for (int off = 16; off > 0; off >>= 1) {
        sp += __shfl_xor(sp, off, 32);
        sf += __shfl_xor(sf, off, 32);
    }
    const float pg = 1.f / (1.f + __expf(-(sp + pw_b[0])));
    const float fg = 1.f / (1.f + __expf(-(sf + fw_b[0])));

    float* po = out + (size_t)b * Dd;
#pragma unroll
    for (int j = 0; j < 32; ++j) {
        const int o = j * 32 + lane;
        po[o] = pg * predv[j] + fg * featv[j];
    }
}

extern "C" void kernel_launch(void* const* d_in, const int* in_sizes, int n_in,
                              void* d_out, int out_size, void* d_ws, size_t ws_size,
                              hipStream_t stream) {
    const float* feature = (const float*)d_in[0];
    const float* prob    = (const float*)d_in[1];
    const float* W       = (const float*)d_in[2];
    const float* bias    = (const float*)d_in[3];
    const float* pw_w    = (const float*)d_in[4];
    const float* pw_b    = (const float*)d_in[5];
    const float* fw_w    = (const float*)d_in[6];
    const float* fw_b    = (const float*)d_in[7];
    float* out = (float*)d_out;

    // ws layout: Wh (16MB packed bf16) | Wl (16MB)
    uint2* Wh = (uint2*)d_ws;
    uint2* Wl = (uint2*)((char*)d_ws + (size_t)Cc * Dd * Dd * sizeof(unsigned short));

    const int nw4 = Cc * Dd * Dd / 4;                       // float4 count = 2M
    wsplit<<<dim3(nw4 / 256), 256, 0, stream>>>(W, Wh, Wl);

    dim3 grid(Bb / BM, Dd / BN);                            // 128 x 8
    expert_gemm<<<grid, TB, 0, stream>>>(feature, prob, Wh, Wl, out);
    gate_epilogue<<<dim3(Bb / 8), 256, 0, stream>>>(out, feature, prob, bias,
                                                    pw_w, pw_b, fw_w, fw_b);
}
